// GPT_28802050687384
// MI455X (gfx1250) — matrix operand order repaired and checked by reading, not compiled, not run on hardware
//
#include <hip/hip_runtime.h>
#include <math.h>

// ---------------------------------------------------------------------------
// Types
// ---------------------------------------------------------------------------
typedef __bf16 bf16_t;
typedef __attribute__((ext_vector_type(16))) __bf16 v16bf;
typedef __attribute__((ext_vector_type(8)))  float  v8f;
typedef __attribute__((ext_vector_type(4)))  int    i4v;   // trivial 16B chunk

union FragU { v16bf v; i4v q[2]; };

__device__ __forceinline__ v8f wmma_bf16(v16bf a, v16bf b, v8f c) {
  // D = A(16x32 bf16) * B(32x16 bf16) + C(16x16 f32)
  return __builtin_amdgcn_wmma_f32_16x16x32_bf16(false, a, false, b,
                                                 (short)0, c, false, false);
}

// CDNA5 async DMA: global -> LDS, 16 bytes per lane, tracked by ASYNCcnt.
// VDST VGPR carries the per-lane LDS byte offset (ISA 08_async_tensor §4.4).
__device__ __forceinline__ void async_b128(uint32_t lds_off, const void* g) {
  asm volatile("global_load_async_to_lds_b128 %0, %1, off"
               :: "v"(lds_off), "v"(g) : "memory");
}
__device__ __forceinline__ void wait_async0() {
  asm volatile("s_wait_asynccnt 0x0" ::: "memory");
}

// A-fragment from LDS tile [16 rows][32 k], pitch 32 bf16.
// ISA 7.12.2 (16-bit A 16x32): lane M = lane&15; lanes<16 hold K {0-7,16-23},
// lanes>=16 hold K {8-15,24-31}; both halves are 2x contiguous 8-element runs.
__device__ __forceinline__ v16bf load_a_frag(const bf16_t* base, int lane) {
  const int m  = lane & 15;
  const int kb = (lane >> 4) * 8;
  FragU u;
  u.q[0] = *(const i4v*)(base + m * 32 + kb);
  u.q[1] = *(const i4v*)(base + m * 32 + 16 + kb);
  return u.v;
}

// B-fragment from LDS tile stored K-major: [16 cols(n)][32 k], pitch 32 bf16.
// lane N = lane&15; lanes<16 hold K {0-7,16-23}, lanes>=16 hold K {8-15,24-31}.
__device__ __forceinline__ v16bf load_b_frag(const bf16_t* base, int lane) {
  const int n  = lane & 15;
  const int kb = (lane >> 4) * 8;
  FragU u;
  u.q[0] = *(const i4v*)(base + n * 32 + kb);
  u.q[1] = *(const i4v*)(base + n * 32 + kb + 16);
  return u.v;
}

// ---------------------------------------------------------------------------
// Generic bf16 GEMM:  C[M][N] = A[M][K] * B[N][K]^T  (+bias)(+relu)(+residual)
// Block tile 128x128, 4 waves of 64x64 (16 WMMAs per wave per 32-k step).
// Tiles are DMA'd with global_load_async_to_lds_b128 into a double-buffered
// LDS stage; next tile's DMA overlaps current tile's WMMAs.
// M must be a multiple of 128; N is guarded (lm_head N=50257): out-of-range
// B rows are clamped at the source — they only feed accumulator columns that
// the epilogue never stores.
// ---------------------------------------------------------------------------
__global__ __launch_bounds__(128)
void gemm_bf16_kernel(const bf16_t* __restrict__ A,
                      const bf16_t* __restrict__ Bw,
                      const float*  __restrict__ bias,
                      const float*  __restrict__ resid,
                      float*        __restrict__ outF,
                      bf16_t*       __restrict__ outB,
                      int M, int N, int K, int relu)
{
  __shared__ bf16_t As[2][128][32];   // [stage][row][k]
  __shared__ bf16_t Bs[2][128][32];

  const int bm   = blockIdx.y * 128;
  const int bn   = blockIdx.x * 128;
  const int t    = threadIdx.x;       // 0..127 : one tile row per thread
  const int lane = t & 31;
  const int wave = t >> 5;
  const int wm   = (wave >> 1) * 64;
  const int wn   = (wave & 1) * 64;
  const int half = lane >> 4;
  const int l15  = lane & 15;

  const int brow   = bn + t;
  const int brow_c = brow < N ? brow : (N - 1);   // clamp (see header comment)

  const bf16_t* aRow = A  + (size_t)(bm + t) * K;
  const bf16_t* bRow = Bw + (size_t)brow_c * K;
  const uint32_t ldsA0 = (uint32_t)(uintptr_t)&As[0][t][0];
  const uint32_t ldsA1 = (uint32_t)(uintptr_t)&As[1][t][0];
  const uint32_t ldsB0 = (uint32_t)(uintptr_t)&Bs[0][t][0];
  const uint32_t ldsB1 = (uint32_t)(uintptr_t)&Bs[1][t][0];

  v8f acc[4][4] = {};

  // Issue one 128x32 A-tile + B-tile (64B per thread each) into stage st.
  auto issue_tile = [&](int k0, int st) {
    const bf16_t* ag = aRow + k0;
    const bf16_t* bg = bRow + k0;
    const uint32_t la = st ? ldsA1 : ldsA0;
    const uint32_t lb = st ? ldsB1 : ldsB0;
    #pragma unroll
    for (int j = 0; j < 4; j++) {
      async_b128(la + 16u * j, ag + 8 * j);
      async_b128(lb + 16u * j, bg + 8 * j);
    }
  };

  int st = 0;
  issue_tile(0, 0);
  wait_async0();
  __syncthreads();

  for (int k0 = 0; k0 < K; k0 += 32) {
    if (k0 + 32 < K) issue_tile(k0 + 32, st ^ 1);   // DMA next tile

    v16bf af[4];
    #pragma unroll
    for (int mi = 0; mi < 4; mi++)
      af[mi] = load_a_frag(&As[st][wm + mi * 16][0], lane);
    #pragma unroll
    for (int ni = 0; ni < 4; ni++) {
      const v16bf bfr = load_b_frag(&Bs[st][wn + ni * 16][0], lane);
      #pragma unroll
      for (int mi = 0; mi < 4; mi++)
        acc[mi][ni] = wmma_bf16(af[mi], bfr, acc[mi][ni]);
    }

    wait_async0();        // next tile landed in LDS
    __syncthreads();      // everyone done reading stage st
    st ^= 1;
  }

  // Epilogue. C/D layout: VGPR i -> row = i + 8*(lane>>4), col = lane&15.
  #pragma unroll
  for (int mi = 0; mi < 4; mi++) {
    #pragma unroll
    for (int ni = 0; ni < 4; ni++) {
      const int col = bn + wn + ni * 16 + l15;
      if (col >= N) continue;
      const float bv   = bias ? bias[col] : 0.0f;
      const int   row0 = bm + wm + mi * 16 + 8 * half;
      #pragma unroll
      for (int i = 0; i < 8; i++) {
        const int row = row0 + i;
        float v = acc[mi][ni][i] + bv;
        if (relu) v = fmaxf(v, 0.0f);
        const size_t off = (size_t)row * N + col;
        if (resid) v += resid[off];
        if (outF) outF[off] = v;
        if (outB) outB[off] = (bf16_t)v;
      }
    }
  }
}

// ---------------------------------------------------------------------------
// Flash-style causal attention. One wave handles a 16-row q tile of one (b,h).
// Q,K: [B*H][T][64] bf16; Vt: [B*H][64][T] bf16 (pre-transposed).
// scale = 1/sqrt(C)  (per reference source, NOT 1/sqrt(HD)).
// ---------------------------------------------------------------------------
__global__ __launch_bounds__(32)
void attn_kernel(const bf16_t* __restrict__ Q, const bf16_t* __restrict__ Kt,
                 const bf16_t* __restrict__ Vt, bf16_t* __restrict__ y,
                 int T, int Hn, float scale)
{
  __shared__ float Ps[16][32];
  const int qtiles = T / 16;
  const int qt   = blockIdx.x % qtiles;
  const int bh   = blockIdx.x / qtiles;
  const int lane = threadIdx.x;
  const int half = lane >> 4;
  const int l15  = lane & 15;
  const int q0   = qt * 16;

  const bf16_t* Qp = Q  + (size_t)bh * T * 64;
  const bf16_t* Kp = Kt + (size_t)bh * T * 64;
  const bf16_t* Vp = Vt + (size_t)bh * 64 * T;

  // Q A-fragments (K = head dim 64 -> two fragments), loaded once.
  v16bf aq[2];
  {
    const int m  = l15;
    const int kb = half * 8;
    #pragma unroll
    for (int g = 0; g < 2; g++) {
      const bf16_t* base = Qp + (size_t)(q0 + m) * 64 + g * 32;
      FragU u;
      u.q[0] = *(const i4v*)(base + kb);
      u.q[1] = *(const i4v*)(base + 16 + kb);
      aq[g] = u.v;
    }
  }

  v8f o[4] = {};                  // O accumulator: 4 n-tiles over HD=64
  float mrow[8], lsum[8];
  #pragma unroll
  for (int i = 0; i < 8; i++) { mrow[i] = -INFINITY; lsum[i] = 0.0f; }

  const int kend = q0 + 16;       // max token index needed = q0+15
  for (int kt = 0; kt < kend; kt += 32) {
    // ---- S(16x32) = Q . K^T over this 32-token window ----
    v8f s[2];
    #pragma unroll
    for (int nt = 0; nt < 2; nt++) {
      v8f c = {};
      #pragma unroll
      for (int g = 0; g < 2; g++) {            // head-dim halves
        const bf16_t* base = Kp + (size_t)(kt + nt * 16 + l15) * 64 + g * 32 + half * 8;
        FragU u;
        u.q[0] = *(const i4v*)(base);
        u.q[1] = *(const i4v*)(base + 16);
        c = wmma_bf16(aq[g], u.v, c);
      }
      s[nt] = c;
    }

    // ---- scale, causal mask, online softmax (16-lane row reductions) ----
    float p0a[8], p1a[8], alpha[8];
    #pragma unroll
    for (int i = 0; i < 8; i++) {
      const int row = q0 + i + 8 * half;
      float v0 = s[0][i] * scale;
      float v1 = s[1][i] * scale;
      if (kt + l15 > row)      v0 = -INFINITY;
      if (kt + 16 + l15 > row) v1 = -INFINITY;
      float mx = fmaxf(v0, v1);
      #pragma unroll
      for (int off = 8; off >= 1; off >>= 1) mx = fmaxf(mx, __shfl_xor(mx, off, 32));
      const float mnew = fmaxf(mrow[i], mx);   // always finite: token kt <= row
      const float a_   = __expf(mrow[i] - mnew);
      const float p0   = __expf(v0 - mnew);
      const float p1   = __expf(v1 - mnew);
      float ls = p0 + p1;
      #pragma unroll
      for (int off = 8; off >= 1; off >>= 1) ls += __shfl_xor(ls, off, 32);
      mrow[i]  = mnew;
      lsum[i]  = lsum[i] * a_ + ls;
      alpha[i] = a_;
      p0a[i] = p0; p1a[i] = p1;
    }
    #pragma unroll
    for (int d = 0; d < 4; d++)
      #pragma unroll
      for (int i = 0; i < 8; i++) o[d][i] *= alpha[i];

    // ---- C-layout P -> A-layout bf16 fragment via LDS ----
    #pragma unroll
    for (int i = 0; i < 8; i++) {
      Ps[i + 8 * half][l15]      = p0a[i];
      Ps[i + 8 * half][16 + l15] = p1a[i];
    }
    __syncthreads();   // single-wave WG: ordering fence (S_NOP barrier + waits)
    v16bf pa;
    {
      const int m  = l15;
      const int kb = half * 8;
      #pragma unroll
      for (int e = 0; e < 8; e++) {
        pa[e]     = (bf16_t)Ps[m][kb + e];
        pa[8 + e] = (bf16_t)Ps[m][16 + kb + e];
      }
    }
    __syncthreads();

    // ---- O += P . V  (V^T is [HD][T], K-major over tokens) ----
    #pragma unroll
    for (int dt = 0; dt < 4; dt++) {
      const bf16_t* base = Vp + (size_t)(dt * 16 + l15) * T + kt + half * 8;
      FragU u;
      u.q[0] = *(const i4v*)(base);
      u.q[1] = *(const i4v*)(base + 16);
      o[dt] = wmma_bf16(pa, u.v, o[dt]);
    }
  }

  // ---- normalize and write y[b][t][h*64+d] (bf16, feeds proj GEMM) ----
  const int C  = Hn * 64;
  const int hh = bh % Hn;
  const int b  = bh / Hn;
  #pragma unroll
  for (int dt = 0; dt < 4; dt++) {
    #pragma unroll
    for (int i = 0; i < 8; i++) {
      const int row = q0 + i + 8 * half;
      const float val = o[dt][i] / lsum[i];
      y[((size_t)(b * T + row)) * C + hh * 64 + dt * 16 + l15] = (bf16_t)val;
    }
  }
}

// ---------------------------------------------------------------------------
// LayerNorm: one wave per row, fp32 in -> bf16 out.
// ---------------------------------------------------------------------------
__global__ __launch_bounds__(128)
void ln_kernel(const float* __restrict__ h, const float* __restrict__ w,
               const float* __restrict__ b, bf16_t* __restrict__ out, int C)
{
  const int row  = blockIdx.x * 4 + (threadIdx.x >> 5);
  const int lane = threadIdx.x & 31;
  const float* x = h + (size_t)row * C;
  float s = 0.0f, s2 = 0.0f;
  for (int c = lane; c < C; c += 32) { const float v = x[c]; s += v; s2 += v * v; }
  #pragma unroll
  for (int off = 16; off >= 1; off >>= 1) {
    s  += __shfl_xor(s,  off, 32);
    s2 += __shfl_xor(s2, off, 32);
  }
  const float mu  = s / C;
  const float var = s2 / C - mu * mu;
  const float rs  = rsqrtf(var + 1e-5f);
  for (int c = lane; c < C; c += 32)
    out[(size_t)row * C + c] = (bf16_t)((x[c] - mu) * rs * w[c] + b[c]);
}

// ---------------------------------------------------------------------------
// Small utility kernels
// ---------------------------------------------------------------------------
__global__ void transpose_cvt_kernel(const float* __restrict__ in,
                                     bf16_t* __restrict__ out,
                                     int Lb, int K, int N)
{ // in: [Lb][K][N] f32 -> out: [Lb][N][K] bf16 (K-major for GEMM B operand)
  const size_t total = (size_t)Lb * K * N;
  for (size_t i = blockIdx.x * (size_t)blockDim.x + threadIdx.x; i < total;
       i += (size_t)gridDim.x * blockDim.x) {
    const size_t l = i / ((size_t)K * N);
    const size_t r = i % ((size_t)K * N);
    const size_t n = r / K;
    const size_t k = r % K;
    out[i] = (bf16_t)in[(l * K + k) * N + n];
  }
}

__global__ void cvt_bf16_kernel(const float* __restrict__ in,
                                bf16_t* __restrict__ out, size_t n)
{
  for (size_t i = blockIdx.x * (size_t)blockDim.x + threadIdx.x; i < n;
       i += (size_t)gridDim.x * blockDim.x)
    out[i] = (bf16_t)in[i];
}

__global__ void embed_kernel(const int* __restrict__ x,
                             const float* __restrict__ wte,
                             const float* __restrict__ wpe,
                             float* __restrict__ h, int BT, int T, int C)
{
  const size_t total = (size_t)BT * C;
  for (size_t i = blockIdx.x * (size_t)blockDim.x + threadIdx.x; i < total;
       i += (size_t)gridDim.x * blockDim.x) {
    const size_t bt = i / C;
    const int    c  = (int)(i % C);
    const int    tk = x[bt];
    const int    tp = (int)(bt % T);
    h[i] = wte[(size_t)tk * C + c] + wpe[(size_t)tp * C + c];
  }
}

__global__ void repack_qkv_kernel(const bf16_t* __restrict__ qkv,
                                  bf16_t* __restrict__ Q, bf16_t* __restrict__ K,
                                  bf16_t* __restrict__ Vt,
                                  int Bn, int T, int Hn, int HDn)
{
  const int C_ = Hn * HDn, C3 = 3 * C_;
  const size_t total = (size_t)Bn * Hn * T * HDn;
  for (size_t i = blockIdx.x * (size_t)blockDim.x + threadIdx.x; i < total;
       i += (size_t)gridDim.x * blockDim.x) {
    const int d = (int)(i % HDn); size_t r = i / HDn;
    const int t = (int)(r % T);   r /= T;
    const int hh = (int)(r % Hn);
    const int b  = (int)(r / Hn);
    const size_t src = ((size_t)(b * T + t)) * C3 + hh * HDn + d;
    const size_t bhT = (size_t)(b * Hn + hh);
    Q [(bhT * T + t) * HDn + d] = qkv[src];
    K [(bhT * T + t) * HDn + d] = qkv[src + C_];
    Vt[(bhT * HDn + d) * T + t] = qkv[src + 2 * C_];
  }
}

// Streaming log-softmax NLL per row; masked mean via atomics.
__global__ __launch_bounds__(256)
void loss_rows_kernel(const float* __restrict__ logits, const int* __restrict__ tgt,
                      float* __restrict__ acc, int V)
{
  const int r = blockIdx.x;
  const float* x = logits + (size_t)r * V;
  float m = -INFINITY, l = 0.0f;
  for (int c = threadIdx.x; c < V; c += 256) {
    const float v = x[c];
    if (v > m) { l = l * __expf(m - v) + 1.0f; m = v; }
    else       { l += __expf(v - m); }
  }
  __shared__ float sm[256], sl[256];
  sm[threadIdx.x] = m; sl[threadIdx.x] = l;
  __syncthreads();
  for (int s = 128; s >= 1; s >>= 1) {
    if (threadIdx.x < s) {
      const float m1 = sm[threadIdx.x], l1 = sl[threadIdx.x];
      const float m2 = sm[threadIdx.x + s], l2 = sl[threadIdx.x + s];
      const float mm = fmaxf(m1, m2);
      sm[threadIdx.x] = mm;
      sl[threadIdx.x] = l1 * __expf(m1 - mm) + l2 * __expf(m2 - mm);
    }
    __syncthreads();
  }
  if (threadIdx.x == 0) {
    const int tg = tgt[r];
    if (tg != 12) {                       // IGNORE = 12
      const float nll = -(x[tg] - sm[0] - __logf(sl[0]));
      atomicAdd(&acc[0], nll);
      atomicAdd(&acc[1], 1.0f);
    }
  }
}

__global__ void zero2_kernel(float* p) { if (threadIdx.x < 2) p[threadIdx.x] = 0.0f; }
__global__ void loss_final_kernel(const float* acc, float* out)
{ *out = acc[0] / fmaxf(acc[1], 1.0f); }

// ---------------------------------------------------------------------------
// Host orchestration
// ---------------------------------------------------------------------------
extern "C" void kernel_launch(void* const* d_in, const int* in_sizes, int n_in,
                              void* d_out, int out_size, void* d_ws, size_t ws_size,
                              hipStream_t stream)
{
  (void)in_sizes; (void)n_in; (void)out_size; (void)ws_size;
  const int Vv = 50257, Cc = 768, Hn = 12, Ll = 12, Bb = 8, Tt = 1024;
  const int Mr = Bb * Tt;                        // 8192 rows
  const float scale = 1.0f / sqrtf((float)Cc);   // reference uses 1/sqrt(C)

  const int*   x     = (const int*)  d_in[0];
  const int*   tgt   = (const int*)  d_in[1];
  const float* wte   = (const float*)d_in[2];
  const float* wpe   = (const float*)d_in[3];
  const float* ln1w  = (const float*)d_in[4];
  const float* ln1b  = (const float*)d_in[5];
  const float* ln2w  = (const float*)d_in[6];
  const float* ln2b  = (const float*)d_in[7];
  const float* qkvw  = (const float*)d_in[8];
  const float* qkvb  = (const float*)d_in[9];
  const float* projw = (const float*)d_in[10];
  const float* projb = (const float*)d_in[11];
  const float* fcw   = (const float*)d_in[12];
  const float* fcb   = (const float*)d_in[13];
  const float* fc2w  = (const float*)d_in[14];
  const float* fc2b  = (const float*)d_in[15];
  const float* lmb   = (const float*)d_in[16];

  char* wsp = (char*)d_ws;
  auto alloc = [&](size_t bytes) -> char* {
    char* p = wsp; wsp += (bytes + 255) & ~(size_t)255; return p;
  };
  bf16_t* wqkvT  = (bf16_t*)alloc((size_t)Ll * 3 * Cc * Cc * sizeof(bf16_t));
  bf16_t* wprojT = (bf16_t*)alloc((size_t)Ll * Cc * Cc * sizeof(bf16_t));
  bf16_t* wfcT   = (bf16_t*)alloc((size_t)Ll * 4 * Cc * Cc * sizeof(bf16_t));
  bf16_t* wfc2T  = (bf16_t*)alloc((size_t)Ll * Cc * 4 * Cc * sizeof(bf16_t));
  bf16_t* wteB   = (bf16_t*)alloc((size_t)Vv * Cc * sizeof(bf16_t));
  float*  h      = (float*) alloc((size_t)Mr * Cc * sizeof(float));
  bf16_t* act    = (bf16_t*)alloc((size_t)Mr * Cc * sizeof(bf16_t));
  bf16_t* ybuf   = (bf16_t*)alloc((size_t)Mr * Cc * sizeof(bf16_t));
  char*   scr    = alloc(2 * (size_t)Mr * 3 * Cc * sizeof(bf16_t)); // qkv+QKV / mlp1
  float*  acc    = (float*) alloc(2 * sizeof(float));

  bf16_t* qkv  = (bf16_t*)scr;
  bf16_t* Qb   = (bf16_t*)(scr + (size_t)Mr * 3 * Cc * sizeof(bf16_t));
  bf16_t* Kb   = Qb + (size_t)Mr * Cc;
  bf16_t* Vb   = Kb + (size_t)Mr * Cc;
  bf16_t* mlp1 = (bf16_t*)scr;   // reuse (qkv/Q/K/V dead by MLP time)

  auto nblk = [](size_t n, int t) { return (unsigned)((n + t - 1) / t); };

  // Weight cast/transpose to K-major bf16 (one-time HBM sweep, ~33us)
  transpose_cvt_kernel<<<nblk((size_t)Ll*Cc*3*Cc,256),256,0,stream>>>(qkvw,  wqkvT,  Ll, Cc,   3*Cc);
  transpose_cvt_kernel<<<nblk((size_t)Ll*Cc*Cc,  256),256,0,stream>>>(projw, wprojT, Ll, Cc,   Cc);
  transpose_cvt_kernel<<<nblk((size_t)Ll*Cc*4*Cc,256),256,0,stream>>>(fcw,   wfcT,   Ll, Cc,   4*Cc);
  transpose_cvt_kernel<<<nblk((size_t)Ll*4*Cc*Cc,256),256,0,stream>>>(fc2w,  wfc2T,  Ll, 4*Cc, Cc);
  cvt_bf16_kernel<<<nblk((size_t)Vv*Cc,256),256,0,stream>>>(wte, wteB, (size_t)Vv * Cc);

  embed_kernel<<<nblk((size_t)Mr*Cc,256),256,0,stream>>>(x, wte, wpe, h, Mr, Tt, Cc);

  auto gemm = [&](const bf16_t* Am, const bf16_t* Bm, const float* bias,
                  const float* resid, float* outF, bf16_t* outB,
                  int M, int N, int K, int relu) {
    dim3 g((N + 127) / 128, M / 128);
    gemm_bf16_kernel<<<g, 128, 0, stream>>>(Am, Bm, bias, resid, outF, outB, M, N, K, relu);
  };

  for (int l = 0; l < Ll; l++) {
    ln_kernel<<<Mr/4, 128, 0, stream>>>(h, ln1w + (size_t)l*Cc, ln1b + (size_t)l*Cc, act, Cc);
    gemm(act, wqkvT + (size_t)l*3*Cc*Cc, qkvb + (size_t)l*3*Cc, nullptr, nullptr, qkv,
         Mr, 3*Cc, Cc, 0);
    repack_qkv_kernel<<<nblk((size_t)Mr*Cc,256),256,0,stream>>>(qkv, Qb, Kb, Vb, Bb, Tt, Hn, 64);
    attn_kernel<<<Bb*Hn*(Tt/16), 32, 0, stream>>>(Qb, Kb, Vb, ybuf, Tt, Hn, scale);
    gemm(ybuf, wprojT + (size_t)l*Cc*Cc, projb + (size_t)l*Cc, h, h, nullptr,
         Mr, Cc, Cc, 0);
    ln_kernel<<<Mr/4, 128, 0, stream>>>(h, ln2w + (size_t)l*Cc, ln2b + (size_t)l*Cc, act, Cc);
    gemm(act, wfcT + (size_t)l*4*Cc*Cc, fcb + (size_t)l*4*Cc, nullptr, nullptr, mlp1,
         Mr, 4*Cc, Cc, 1);
    gemm(mlp1, wfc2T + (size_t)l*Cc*4*Cc, fc2b + (size_t)l*Cc, h, h, nullptr,
         Mr, Cc, 4*Cc, 0);
  }

  // Tied lm_head: logits = h @ wte^T + lm_b   (wte is [V][C] = K-major already)
  cvt_bf16_kernel<<<nblk((size_t)Mr*Cc,256),256,0,stream>>>(h, act, (size_t)Mr * Cc);
  float* logits = (float*)d_out;
  gemm(act, wteB, lmb, nullptr, logits, nullptr, Mr, Vv, Cc, 0);

  // Loss -> d_out[B*T*V]
  zero2_kernel<<<1, 32, 0, stream>>>(acc);
  loss_rows_kernel<<<Mr, 256, 0, stream>>>(logits, tgt, acc, Vv);
  loss_final_kernel<<<1, 1, 0, stream>>>(acc, logits + (size_t)Mr * Vv);
}
